// GeneratorModel_9423158248268
// MI455X (gfx1250) — matrix operand order repaired and checked by reading, not compiled, not run on hardware
//
#include <hip/hip_runtime.h>
#include <cstdint>
#include <cstddef>

// ---------------------------------------------------------------------------
// Types for CDNA5 WMMA (wave32): v_wmma_f32_16x16x32_bf16
// ---------------------------------------------------------------------------
typedef __bf16 bf16_t;
typedef __bf16 v16bf __attribute__((ext_vector_type(16)));
typedef float  v8f   __attribute__((ext_vector_type(8)));

#define DEV __device__ __forceinline__

static constexpr int V     = 32000;
static constexpr int E     = 64;
static constexpr int H     = 512;
static constexpr int G4    = 4 * H;      // 2048 gate width
static constexpr int B     = 32;
static constexpr int S_CTX = 64;
static constexpr int T_DEC = 32;         // S_OUT - 1
static constexpr int ROWS  = T_DEC * B;  // 1024 decoder rows
static constexpr int D2    = 2 * H;      // 1024 (h_dec ++ ctx_vec)
static constexpr int CHUNK = 128;        // logits rows per chunk (16 MB scratch)

DEV bf16_t to_bf16(float f) { return (bf16_t)f; }
DEV float  sigmoidf(float x) { return 1.f / (1.f + __expf(-x)); }

// A-tile (16x32 bf16), ISA 7.12.2 layout:
//   lanes 0-15 : row r0+l,     K blocks [k0, k0+8) and [k0+16, k0+24)
//   lanes16-31 : row r0+l-16,  K blocks [k0+8,k0+16) and [k0+24, k0+32)
DEV v16bf load_a_tile(const bf16_t* A, int lda, int r0, int k0, int lane) {
  int row = r0 + (lane & 15);
  int kb  = (lane < 16) ? 0 : 8;
  const bf16_t* p = A + (size_t)row * lda + k0 + kb;
  union { uint4 u[2]; v16bf v; } t;
  t.u[0] = *reinterpret_cast<const uint4*>(p);
  t.u[1] = *reinterpret_cast<const uint4*>(p + 16);
  return t.v;
}

// B-tile (32x16 bf16) from N-major (pre-transposed) weights BT[N][K]:
//   lanes 0-15 : col n0+l, K [k0, k0+8),[k0+16,k0+24) ; lanes 16-31 : col, K [k0+8, k0+16),[k0+24,k0+32)
DEV v16bf load_bt_tile(const bf16_t* BT, int ldk, int n0, int k0, int lane) {
  int col = n0 + (lane & 15);
  int kb  = (lane < 16) ? 0 : 8;
  const bf16_t* p = BT + (size_t)col * ldk + k0 + kb;
  union { uint4 u[2]; v16bf v; } t;
  t.u[0] = *reinterpret_cast<const uint4*>(p);
  t.u[1] = *reinterpret_cast<const uint4*>(p + 16);
  return t.v;
}

DEV v8f wmma_bf16(v16bf a, v16bf b, v8f c) {
  return __builtin_amdgcn_wmma_f32_16x16x32_bf16(false, a, false, b, (short)0, c,
                                                 false, false);
}

// ---------------------------------------------------------------------------
// Elementwise prep kernels
// ---------------------------------------------------------------------------
__global__ void transpose_f32_bf16_kernel(const float* __restrict__ W,
                                          bf16_t* __restrict__ WT,
                                          int K, int N, int total) {
  for (int i = blockIdx.x * blockDim.x + threadIdx.x; i < total;
       i += gridDim.x * blockDim.x) {
    int k = i / N, n = i - k * N;
    WT[(size_t)n * K + k] = to_bf16(W[i]);
  }
}

__global__ void gather_embed_kernel(const int* __restrict__ tok,
                                    const float* __restrict__ emb,
                                    bf16_t* __restrict__ x, int total) {
  // E == 64 hardcoded via shifts
  for (int i = blockIdx.x * blockDim.x + threadIdx.x; i < total;
       i += gridDim.x * blockDim.x) {
    int row = i >> 6, e = i & 63;
    x[i] = to_bf16(emb[(size_t)tok[row] * E + e]);
  }
}

__global__ void zero_state_kernel(bf16_t* h, float* c, int n) {
  int i = blockIdx.x * blockDim.x + threadIdx.x;
  if (i < n) { h[i] = to_bf16(0.f); c[i] = 0.f; }
}

// ---------------------------------------------------------------------------
// Fused LSTM step: gates = x_t@Wih + h@Whh + b  (bf16 WMMA, f32 acc),
// then cell update. One wave per 16 columns of H; each block owns all
// four gates for its columns so the i/f/g/o coupling is local.
// grid = H/16 blocks, block = 32 threads (one wave, EXEC all-ones).
// ---------------------------------------------------------------------------
__global__ void lstm_step_kernel(const bf16_t* __restrict__ x_t,    // [B,E]
                                 const bf16_t* __restrict__ h_prev, // [B,H]
                                 const float*  __restrict__ c_prev, // [B,H]
                                 const bf16_t* __restrict__ wihT,   // [4H,E]
                                 const bf16_t* __restrict__ whhT,   // [4H,H]
                                 const float*  __restrict__ bias,   // [4H]
                                 bf16_t* __restrict__ h_new,        // [B,H]
                                 float*  __restrict__ c_new,        // [B,H]
                                 float*  __restrict__ h_f32,        // opt [B,H]
                                 bf16_t* __restrict__ h_bf16_s,     // opt [B,s_ld]
                                 int s_ld) {
  const int lane = threadIdx.x;
  const int n0   = blockIdx.x * 16;

  v8f acc[2][4] = {};  // [m-tile][gate]

  // input contribution, K = E = 64
  for (int k0 = 0; k0 < E; k0 += 32) {
    v16bf a0 = load_a_tile(x_t, E, 0,  k0, lane);
    v16bf a1 = load_a_tile(x_t, E, 16, k0, lane);
#pragma unroll
    for (int g = 0; g < 4; ++g) {
      v16bf b = load_bt_tile(wihT, E, g * H + n0, k0, lane);
      acc[0][g] = wmma_bf16(a0, b, acc[0][g]);
      acc[1][g] = wmma_bf16(a1, b, acc[1][g]);
    }
  }
  // recurrent contribution, K = H = 512
  for (int k0 = 0; k0 < H; k0 += 32) {
    v16bf a0 = load_a_tile(h_prev, H, 0,  k0, lane);
    v16bf a1 = load_a_tile(h_prev, H, 16, k0, lane);
#pragma unroll
    for (int g = 0; g < 4; ++g) {
      v16bf b = load_bt_tile(whhT, H, g * H + n0, k0, lane);
      acc[0][g] = wmma_bf16(a0, b, acc[0][g]);
      acc[1][g] = wmma_bf16(a1, b, acc[1][g]);
    }
  }

  const int col = n0 + (lane & 15);
  const float bi = bias[0 * H + col], bff = bias[1 * H + col];
  const float bg = bias[2 * H + col], bo  = bias[3 * H + col];
#pragma unroll
  for (int m = 0; m < 2; ++m) {
    int rbase = m * 16 + ((lane < 16) ? 0 : 8);
#pragma unroll
    for (int i = 0; i < 8; ++i) {
      int row = rbase + i;
      float ig = sigmoidf(acc[m][0][i] + bi);
      float fg = sigmoidf(acc[m][1][i] + bff);
      float gg = tanhf   (acc[m][2][i] + bg);
      float og = sigmoidf(acc[m][3][i] + bo);
      float c  = fg * c_prev[(size_t)row * H + col] + ig * gg;
      float h  = og * tanhf(c);
      c_new[(size_t)row * H + col] = c;
      h_new[(size_t)row * H + col] = to_bf16(h);
      if (h_f32)    h_f32[(size_t)row * H + col] = h;
      if (h_bf16_s) h_bf16_s[(size_t)row * s_ld + col] = to_bf16(h);
    }
  }
}

// ---------------------------------------------------------------------------
// Generic C[M,N] = A[M,K](bf16,row-major) @ BT[N,K](bf16) + bias, f32 out.
// One wave computes a 32x16 tile (2 M-tiles reuse one B tile).
// grid = (N/16, M/32), block = 32.
// ---------------------------------------------------------------------------
__global__ void gemm_bf16_tn_kernel(const bf16_t* __restrict__ A, int lda,
                                    const bf16_t* __restrict__ BT, int ldb,
                                    const float* __restrict__ bias,
                                    float* __restrict__ C, int ldc, int K) {
  const int lane = threadIdx.x;
  const int n0 = blockIdx.x * 16;
  const int r0 = blockIdx.y * 32;
  v8f acc0 = {}, acc1 = {};
  for (int k0 = 0; k0 < K; k0 += 32) {
    v16bf a0 = load_a_tile(A, lda, r0,      k0, lane);
    v16bf a1 = load_a_tile(A, lda, r0 + 16, k0, lane);
    v16bf b  = load_bt_tile(BT, ldb, n0, k0, lane);
    if (k0 + 128 < K)  // stream next B tiles through L2 (global_prefetch_b8)
      __builtin_prefetch(BT + (size_t)(n0 + (lane & 15)) * ldb + k0 + 128, 0, 1);
    acc0 = wmma_bf16(a0, b, acc0);
    acc1 = wmma_bf16(a1, b, acc1);
  }
  const int col = n0 + (lane & 15);
  const float bv = bias ? bias[col] : 0.f;
  const int rb = r0 + ((lane < 16) ? 0 : 8);
#pragma unroll
  for (int i = 0; i < 8; ++i) {
    C[(size_t)(rb + i) * ldc + col]      = acc0[i] + bv;
    C[(size_t)(rb + 16 + i) * ldc + col] = acc1[i] + bv;
  }
}

// ---------------------------------------------------------------------------
// Attention: per (t,b) row — scores over S_CTX, mask PAD, softmax,
// context vector written (bf16) into concat[:, 512:1024]; attn saved for copy.
// ---------------------------------------------------------------------------
__global__ void attention_kernel(const float* __restrict__ q,    // [ROWS,H]
                                 const float* __restrict__ enc,  // [S,B,H]
                                 const int*   __restrict__ ctx,  // [S,B]
                                 float*       __restrict__ attn, // [ROWS,S]
                                 bf16_t*      __restrict__ concat) {
  const int row = blockIdx.x;
  const int b   = row % B;
  const int tid = threadIdx.x;
  __shared__ float qs[H];
  __shared__ float sc[S_CTX];
  for (int i = tid; i < H; i += blockDim.x) qs[i] = q[(size_t)row * H + i];
  __syncthreads();
  if (tid < S_CTX) {
    const float* e = enc + ((size_t)tid * B + b) * H;
    float d = 0.f;
    for (int k = 0; k < H; ++k) d += qs[k] * e[k];
    if (ctx[tid * B + b] == 0) d = -1e9f;   // PAD mask
    sc[tid] = d;
  }
  __syncthreads();
  if (tid == 0) {
    float m = sc[0];
    for (int s = 1; s < S_CTX; ++s) m = fmaxf(m, sc[s]);
    float sum = 0.f;
    for (int s = 0; s < S_CTX; ++s) { float p = __expf(sc[s] - m); sc[s] = p; sum += p; }
    float inv = 1.f / sum;
    for (int s = 0; s < S_CTX; ++s) sc[s] *= inv;
  }
  __syncthreads();
  if (tid < S_CTX) attn[(size_t)row * S_CTX + tid] = sc[tid];
  for (int c = tid; c < H; c += blockDim.x) {
    float a = 0.f;
    for (int s = 0; s < S_CTX; ++s) a += sc[s] * enc[((size_t)s * B + b) * H + c];
    concat[(size_t)row * D2 + H + c] = to_bf16(a);
  }
}

// copy mechanism: logits[row, ctx[s,b]] += attn[row, s]
__global__ void copy_scatter_kernel(float* __restrict__ logits,      // [CHUNK,V]
                                    const float* __restrict__ attn,  // [ROWS,S]
                                    const int* __restrict__ ctx, int r0, int rows) {
  int i = blockIdx.x * blockDim.x + threadIdx.x;
  if (i >= rows * S_CTX) return;
  int lrow = i >> 6, s = i & 63;
  int row = r0 + lrow;
  int b = row & (B - 1);
  int v = ctx[s * B + b];
  atomicAdd(&logits[(size_t)lrow * V + v], attn[(size_t)row * S_CTX + s]);
}

// per-row log-softmax + NLL at target; mask = (tgt != PAD)
__global__ void nll_kernel(const float* __restrict__ logits, // [CHUNK,V]
                           const int* __restrict__ out_tok,  // [S_OUT,B]
                           int r0, float* __restrict__ nll, float* __restrict__ maskv) {
  const int lrow = blockIdx.x;
  const int row = r0 + lrow;
  const int t = row / B, b = row % B;
  const int tgt = out_tok[(t + 1) * B + b];
  const float* lr = logits + (size_t)lrow * V;
  __shared__ float red[256];
  const int tid = threadIdx.x;
  float m = -3.4e38f;
  for (int v = tid; v < V; v += 256) m = fmaxf(m, lr[v]);
  red[tid] = m; __syncthreads();
  for (int s = 128; s > 0; s >>= 1) {
    if (tid < s) red[tid] = fmaxf(red[tid], red[tid + s]);
    __syncthreads();
  }
  m = red[0]; __syncthreads();
  float sum = 0.f;
  for (int v = tid; v < V; v += 256) sum += __expf(lr[v] - m);
  red[tid] = sum; __syncthreads();
  for (int s = 128; s > 0; s >>= 1) {
    if (tid < s) red[tid] += red[tid + s];
    __syncthreads();
  }
  if (tid == 0) {
    float lse = m + __logf(red[0]);
    float n = lse - lr[tgt];
    nll[row]   = (tgt != 0) ? n : 0.f;
    maskv[row] = (tgt != 0) ? 1.f : 0.f;
  }
}

__global__ void loss_kernel(const float* __restrict__ nll,
                            const float* __restrict__ maskv, float* out) {
  __shared__ float rn[256], rm[256];
  const int tid = threadIdx.x;
  float sn = 0.f, sm = 0.f;
  for (int i = tid; i < ROWS; i += 256) { sn += nll[i]; sm += maskv[i]; }
  rn[tid] = sn; rm[tid] = sm; __syncthreads();
  for (int s = 128; s > 0; s >>= 1) {
    if (tid < s) { rn[tid] += rn[tid + s]; rm[tid] += rm[tid + s]; }
    __syncthreads();
  }
  if (tid == 0) out[0] = rn[0] / fmaxf(rm[0], 1.f);
}

// ---------------------------------------------------------------------------
// Host orchestration
// ---------------------------------------------------------------------------
extern "C" void kernel_launch(void* const* d_in, const int* in_sizes, int n_in,
                              void* d_out, int out_size, void* d_ws, size_t ws_size,
                              hipStream_t stream) {
  (void)in_sizes; (void)n_in; (void)out_size; (void)ws_size;
  const int*   ctx      = (const int*)  d_in[0];
  const int*   out_tok  = (const int*)  d_in[1];
  const float* emb_enc  = (const float*)d_in[2];
  const float* w_ih_enc = (const float*)d_in[3];
  const float* w_hh_enc = (const float*)d_in[4];
  const float* b_enc    = (const float*)d_in[5];
  const float* emb_dec  = (const float*)d_in[6];
  const float* w_ih_dec = (const float*)d_in[7];
  const float* w_hh_dec = (const float*)d_in[8];
  const float* b_dec    = (const float*)d_in[9];
  const float* w_q      = (const float*)d_in[10];
  const float* w_out    = (const float*)d_in[11];
  const float* b_out    = (const float*)d_in[12];
  float* loss_out = (float*)d_out;

  char* wp = (char*)d_ws;
  auto alloc = [&](size_t bytes) -> char* {
    char* p = wp;
    wp += (bytes + 255) & ~(size_t)255;
    return p;
  };
  bf16_t* x_enc  = (bf16_t*)alloc((size_t)S_CTX * B * E * 2);
  bf16_t* x_dec  = (bf16_t*)alloc((size_t)T_DEC * B * E * 2);
  bf16_t* wihT_e = (bf16_t*)alloc((size_t)G4 * E * 2);
  bf16_t* whhT_e = (bf16_t*)alloc((size_t)G4 * H * 2);
  bf16_t* wihT_d = (bf16_t*)alloc((size_t)G4 * E * 2);
  bf16_t* whhT_d = (bf16_t*)alloc((size_t)G4 * H * 2);
  bf16_t* wqT    = (bf16_t*)alloc((size_t)H * H * 2);
  bf16_t* woutT  = (bf16_t*)alloc((size_t)V * D2 * 2);     // 64 MB, L2-resident
  bf16_t* hbuf0  = (bf16_t*)alloc((size_t)B * H * 2);
  bf16_t* hbuf1  = (bf16_t*)alloc((size_t)B * H * 2);
  float*  cbuf0  = (float*) alloc((size_t)B * H * 4);
  float*  cbuf1  = (float*) alloc((size_t)B * H * 4);
  float*  enc    = (float*) alloc((size_t)S_CTX * B * H * 4);
  float*  qbuf   = (float*) alloc((size_t)ROWS * H * 4);
  bf16_t* concat = (bf16_t*)alloc((size_t)ROWS * D2 * 2);
  float*  attn   = (float*) alloc((size_t)ROWS * S_CTX * 4);
  float*  logits = (float*) alloc((size_t)CHUNK * V * 4);  // 16 MB chunk
  float*  nll    = (float*) alloc((size_t)ROWS * 4);
  float*  maskv  = (float*) alloc((size_t)ROWS * 4);

  const int thr = 256;
  auto gs = [](int n) { return (n + 255) / 256; };

  // weight transpose + bf16 convert (WMMA B-operand wants N-major contiguous K)
  transpose_f32_bf16_kernel<<<gs(E * G4), thr, 0, stream>>>(w_ih_enc, wihT_e, E, G4, E * G4);
  transpose_f32_bf16_kernel<<<gs(H * G4), thr, 0, stream>>>(w_hh_enc, whhT_e, H, G4, H * G4);
  transpose_f32_bf16_kernel<<<gs(E * G4), thr, 0, stream>>>(w_ih_dec, wihT_d, E, G4, E * G4);
  transpose_f32_bf16_kernel<<<gs(H * G4), thr, 0, stream>>>(w_hh_dec, whhT_d, H, G4, H * G4);
  transpose_f32_bf16_kernel<<<gs(H * H),  thr, 0, stream>>>(w_q, wqT, H, H, H * H);
  transpose_f32_bf16_kernel<<<gs(D2 * V), thr, 0, stream>>>(w_out, woutT, D2, V, D2 * V);

  // embeddings
  gather_embed_kernel<<<gs(S_CTX * B * E), thr, 0, stream>>>(ctx, emb_enc, x_enc, S_CTX * B * E);
  gather_embed_kernel<<<gs(T_DEC * B * E), thr, 0, stream>>>(out_tok, emb_dec, x_dec, T_DEC * B * E);

  // h0 = c0 = 0
  zero_state_kernel<<<gs(B * H), thr, 0, stream>>>(hbuf0, cbuf0, B * H);

  // encoder: 64 sequential steps, ping-pong h/c
  for (int t = 0; t < S_CTX; ++t) {
    const bf16_t* hp = (t & 1) ? hbuf1 : hbuf0;
    bf16_t*       hn = (t & 1) ? hbuf0 : hbuf1;
    const float*  cp = (t & 1) ? cbuf1 : cbuf0;
    float*        cn = (t & 1) ? cbuf0 : cbuf1;
    lstm_step_kernel<<<H / 16, 32, 0, stream>>>(
        x_enc + (size_t)t * B * E, hp, cp, wihT_e, whhT_e, b_enc,
        hn, cn, enc + (size_t)t * B * H, nullptr, 0);
  }
  // decoder: continues from final encoder state (S_CTX even -> same parity)
  for (int t = 0; t < T_DEC; ++t) {
    const bf16_t* hp = (t & 1) ? hbuf1 : hbuf0;
    bf16_t*       hn = (t & 1) ? hbuf0 : hbuf1;
    const float*  cp = (t & 1) ? cbuf1 : cbuf0;
    float*        cn = (t & 1) ? cbuf0 : cbuf1;
    lstm_step_kernel<<<H / 16, 32, 0, stream>>>(
        x_dec + (size_t)t * B * E, hp, cp, wihT_d, whhT_d, b_dec,
        hn, cn, nullptr, concat + (size_t)t * B * D2, D2);
  }

  // q = h_dec @ w_q   (A = concat[:, :512], bf16)
  gemm_bf16_tn_kernel<<<dim3(H / 16, ROWS / 32), 32, 0, stream>>>(
      concat, D2, wqT, H, nullptr, qbuf, H, H);

  // attention -> attn + concat[:, 512:]
  attention_kernel<<<ROWS, 256, 0, stream>>>(qbuf, enc, ctx, attn, concat);

  // output projection (dominant GEMM) chunked + copy scatter + NLL
  for (int r0 = 0; r0 < ROWS; r0 += CHUNK) {
    gemm_bf16_tn_kernel<<<dim3(V / 16, CHUNK / 32), 32, 0, stream>>>(
        concat + (size_t)r0 * D2, D2, woutT, D2, b_out, logits, V, D2);
    copy_scatter_kernel<<<gs(CHUNK * S_CTX), thr, 0, stream>>>(logits, attn, ctx, r0, CHUNK);
    nll_kernel<<<CHUNK, 256, 0, stream>>>(logits, out_tok, r0, nll, maskv);
  }

  loss_kernel<<<1, 256, 0, stream>>>(nll, maskv, loss_out);
}
